// MHCEncoderBlock_18537078849766
// MI455X (gfx1250) — matrix operand order repaired and checked by reading, not compiled, not run on hardware
//
#include <hip/hip_runtime.h>

typedef __attribute__((ext_vector_type(16))) _Float16 v16h;
typedef __attribute__((ext_vector_type(8)))  float    v8f;
typedef __attribute__((ext_vector_type(4)))  unsigned int v4u;
typedef __attribute__((ext_vector_type(4)))  int v4i;
typedef __attribute__((ext_vector_type(8)))  int v8i;

#define B_    32
#define T_    1024
#define NS    4
#define D_    192
#define ND    768
#define MLP_  768
#define HEADS 3
#define HD_   64
#define MT    (B_*T_)   /* 32768 tokens */

// ---------------------------------------------------------------- wave helpers
__device__ __forceinline__ float wsum(float v) {
#pragma unroll
  for (int m = 16; m >= 1; m >>= 1) v += __shfl_xor(v, m, 32);
  return v;
}

// ---------------------------------------------------------------- WMMA frag loaders
// A fragment: 16(M) x 32(K) f16.  lane&15 = M row; lane>>4 selects K 16-block.
// element i -> K = 16*hi + i
__device__ __forceinline__ v16h load_a(const _Float16* A, int ld, int row0, int k0) {
  const int lane = threadIdx.x & 31, lh = lane & 15, hi = lane >> 4;
  const _Float16* p = A + (size_t)(row0 + lh) * ld + k0 + hi * 16;
  v16h f;
#pragma unroll
  for (int i = 0; i < 8; ++i) f[i] = p[i];
#pragma unroll
  for (int i = 0; i < 8; ++i) f[8 + i] = p[8 + i];
  return f;
}

// B fragment: 32(K) x 16(N) f16, element i <-> K = 16*hi + i, N = lane&15.
// Source laid out transposed [N][K] row-major -> contiguous 32-byte reads.
__device__ __forceinline__ v16h load_bT(const _Float16* Bt, int ld, int n0, int k0) {
  const int lane = threadIdx.x & 31, lh = lane & 15, hi = lane >> 4;
  const _Float16* p = Bt + (size_t)(n0 + lh) * ld + k0 + hi * 16;
  v16h f;
#pragma unroll
  for (int i = 0; i < 16; ++i) f[i] = p[i];
  return f;
}

// B fragment from [K][N] row-major source (strided reads; used for V in LDS).
__device__ __forceinline__ v16h load_b(const _Float16* Bm, int ld, int k0, int col0) {
  const int lane = threadIdx.x & 31, lh = lane & 15, hi = lane >> 4;
  const _Float16* p = Bm + (size_t)(k0 + hi * 16) * ld + col0 + lh;
  v16h f;
#pragma unroll
  for (int i = 0; i < 16; ++i) f[i] = p[(size_t)i * ld];
  return f;
}

// ---------------------------------------------------------------- Tensor Data Mover
// 2D f16 tile DMA: rows x rowlen elements, row stride in elements, into LDS.
// D# packing per CDNA5 ISA ch.8 (group0: count/lds/global/type, group1: dims).
// Generic pointers to __shared__ carry the LDS byte offset in their low 32 bits
// (flat-aperture rule, ISA 10.2), which is exactly D#.lds_addr.
__device__ __forceinline__ void tdm_load_2d_f16(const _Float16* gsrc, void* lds_dst,
                                                int rows, int rowlen, int row_stride) {
  unsigned long long ga = (unsigned long long)(uintptr_t)gsrc;
  unsigned lds_off = (unsigned)(uintptr_t)lds_dst;
  v4u g0;
  g0[0] = 1u;                                   // count=1, user descriptor
  g0[1] = lds_off;                              // lds_addr (bytes)
  g0[2] = (unsigned)(ga & 0xffffffffu);         // global_addr[31:0]
  g0[3] = (unsigned)((ga >> 32) & 0x01ffffffu)  // global_addr[56:32]
          | (2u << 30);                         // type=2 ("image")
  v8i g1;
  g1[0] = (int)(1u << 16);                      // wg_mask=0 | data_size=1 (2B)
  g1[1] = (int)((unsigned)rowlen << 16);        // tensor_dim0[15:0]
  g1[2] = (int)((unsigned)rows << 16);          // tensor_dim0[31:16]=0 | tensor_dim1[15:0]
  g1[3] = (int)((unsigned)rowlen << 16);        // tensor_dim1[31:16]=0 | tile_dim0
  g1[4] = rows;                                 // tile_dim1 | tile_dim2=0
  g1[5] = row_stride;                           // tensor_dim0_stride[31:0]
  g1[6] = 0;                                    // stride hi | tensor_dim1_stride lo
  g1[7] = 0;
  v4i gz = {0, 0, 0, 0};
#if defined(__clang_major__) && (__clang_major__ >= 23)
  v8i gz8 = {0, 0, 0, 0, 0, 0, 0, 0};
  __builtin_amdgcn_tensor_load_to_lds(g0, g1, gz, gz, gz8, 0);
#else
  __builtin_amdgcn_tensor_load_to_lds(g0, g1, gz, gz, 0);
#endif
}

// ---------------------------------------------------------------- weight convert (f32 [K][N] -> f16 [N][K])
__global__ __launch_bounds__(256) void convert_t_kernel(const float* __restrict__ w,
                                                        _Float16* __restrict__ wt,
                                                        int K, int N) {
  int i = blockIdx.x * 256 + threadIdx.x;
  if (i < K * N) {
    int k = i / N, n = i - k * N;
    wt[(size_t)n * K + k] = (_Float16)w[i];
  }
}

// ---------------------------------------------------------------- gate kernel
// one wave per token: RMSNorm, 24 matvec outputs, sigmoid gates, Sinkhorn(4x4),
// x_res -> d_out, x_in (f32) + fused LN1 -> f16
__global__ __launch_bounds__(256) void gate_kernel(
    const float* __restrict__ x, const float* __restrict__ nw,
    const float* __restrict__ phi_pre, const float* __restrict__ phi_post,
    const float* __restrict__ phi_res, const float* __restrict__ b_pre,
    const float* __restrict__ b_post, const float* __restrict__ b_res,
    const float* ap, const float* apo, const float* ar,
    const float* __restrict__ ln1w, const float* __restrict__ ln1b,
    float* __restrict__ xres_out, float* __restrict__ xin_out,
    _Float16* __restrict__ h1_out, float* __restrict__ hpost_out) {
  __shared__ float xs[8][768];
  __shared__ float hres_s[8][16];
  __shared__ float hpre_s[8][4];
  const int wv = threadIdx.x >> 5;
  const int lane = threadIdx.x & 31;
  const int token = blockIdx.x * 8 + wv;
  const float* xt = x + (size_t)token * ND;

  float xr[24];
  float ssq = 0.f;
#pragma unroll
  for (int k = 0; k < 24; ++k) {
    float v = xt[lane + 32 * k];
    xr[k] = v;
    xs[wv][lane + 32 * k] = v;
    ssq += v * v;
  }
  float irms = rsqrtf(wsum(ssq) * (1.f / 768.f) + 1e-8f);
  float st[24];
#pragma unroll
  for (int k = 0; k < 24; ++k) st[k] = xr[k] * irms * nw[lane + 32 * k];

  const float apre = ap[0], apost = apo[0], ares = ar[0];

  for (int c = 0; c < 4; ++c) {
    float sp = 0.f, sq = 0.f;
#pragma unroll
    for (int k = 0; k < 24; ++k) {
      int e = lane + 32 * k;
      sp += st[k] * phi_pre[e * 4 + c];
      sq += st[k] * phi_post[e * 4 + c];
    }
    sp = wsum(sp);
    sq = wsum(sq);
    if (lane == 0) {
      hpre_s[wv][c] = 1.f / (1.f + __expf(-(apre * sp + b_pre[c])));
      hpost_out[(size_t)token * 4 + c] = 2.f / (1.f + __expf(-(apost * sq + b_post[c])));
    }
  }

  // res logits -> lanes 0..15 (lane = i*4+j)
  float mres = -1e30f;
  for (int cc = 0; cc < 16; ++cc) {
    float s = 0.f;
#pragma unroll
    for (int k = 0; k < 24; ++k) s += st[k] * phi_res[(lane + 32 * k) * 16 + cc];
    s = wsum(s);
    if (lane == cc) mres = ares * s + b_res[cc];
  }
  // Sinkhorn: global max-shift, exp, 20x row/col normalization via lane shuffles
  float g = mres;
#pragma unroll
  for (int m = 8; m >= 1; m >>= 1) g = fmaxf(g, __shfl_xor(g, m, 16));
  float hm = __expf(mres - g);
  for (int it = 0; it < 20; ++it) {
    float rs = hm + __shfl_xor(hm, 1, 4) + __shfl_xor(hm, 2, 4);   // sum over j
    hm = hm / (rs + 1e-8f);
    float cs = hm + __shfl_xor(hm, 4, 16) + __shfl_xor(hm, 8, 16); // sum over i
    hm = hm / (cs + 1e-8f);
  }
  if (lane < 16) hres_s[wv][lane] = hm;
  __syncthreads();

  // x_res = h_res @ x   (per token: 4x4 times 4x192)
#pragma unroll
  for (int k = 0; k < 24; ++k) {
    int e = lane + 32 * k;
    int i = e / 192, dd = e - i * 192;
    float s = 0.f;
#pragma unroll
    for (int j = 0; j < 4; ++j) s += hres_s[wv][i * 4 + j] * xs[wv][j * 192 + dd];
    xres_out[(size_t)token * ND + e] = s;
  }

  // x_in = sum_n h_pre[n] * x[n,:]; fused LayerNorm1 -> f16
  float xin[6];
  float msum = 0.f;
#pragma unroll
  for (int k = 0; k < 6; ++k) {
    int dd = lane + 32 * k;
    float s = 0.f;
#pragma unroll
    for (int n = 0; n < 4; ++n) s += hpre_s[wv][n] * xs[wv][n * 192 + dd];
    xin[k] = s;
    msum += s;
    xin_out[(size_t)token * D_ + dd] = s;
  }
  float mean = wsum(msum) * (1.f / 192.f);
  float vs = 0.f;
#pragma unroll
  for (int k = 0; k < 6; ++k) {
    float d = xin[k] - mean;
    vs += d * d;
  }
  float rstd = rsqrtf(wsum(vs) * (1.f / 192.f) + 1e-5f);
#pragma unroll
  for (int k = 0; k < 6; ++k) {
    int dd = lane + 32 * k;
    h1_out[(size_t)token * D_ + dd] = (_Float16)((xin[k] - mean) * rstd * ln1w[dd] + ln1b[dd]);
  }
}

// ---------------------------------------------------------------- WMMA GEMM
// C[M,N] = A[M,K](f16) * B[K,N] (B given transposed [N][K] f16), fp32 accum.
// Block: 8 waves, each wave 16x64; block tile 128 x 64.
#define EPI_F16 0
#define EPI_RES 1
#define EPI_GELU 2

template <int EPI>
__global__ __launch_bounds__(256) void gemm_kernel(const _Float16* __restrict__ A,
                                                   const _Float16* __restrict__ Bt,
                                                   void* __restrict__ outp,
                                                   const float* __restrict__ res,
                                                   int M, int N, int K) {
  const int wv = threadIdx.x >> 5;
  const int lane = threadIdx.x & 31, lh = lane & 15, hi = lane >> 4;
  const int m0 = blockIdx.x * 128 + wv * 16;
  const int n0 = blockIdx.y * 64;
  (void)M;
  v8f zero = {0.f, 0.f, 0.f, 0.f, 0.f, 0.f, 0.f, 0.f};
  v8f acc[4] = {zero, zero, zero, zero};
  for (int k0 = 0; k0 < K; k0 += 32) {
    v16h a = load_a(A, K, m0, k0);
#pragma unroll
    for (int j = 0; j < 4; ++j) {
      v16h bf = load_bT(Bt, K, n0 + 16 * j, k0);
      acc[j] = __builtin_amdgcn_wmma_f32_16x16x32_f16(false, a, false, bf, (short)0,
                                                      acc[j], false, false);
    }
  }
#pragma unroll
  for (int j = 0; j < 4; ++j) {
    int col = n0 + 16 * j + lh;
#pragma unroll
    for (int r = 0; r < 8; ++r) {
      size_t idx = (size_t)(m0 + hi * 8 + r) * N + col;
      float v = acc[j][r];
      if (EPI == EPI_F16) {
        ((_Float16*)outp)[idx] = (_Float16)v;
      } else if (EPI == EPI_RES) {
        ((float*)outp)[idx] = v + res[idx];
      } else {  // tanh-approx GELU (jax.nn.gelu default), store f16
        float gg = 0.5f * v * (1.f + tanhf(0.7978845608028654f * (v + 0.044715f * v * v * v)));
        ((_Float16*)outp)[idx] = (_Float16)gg;
      }
    }
  }
}

// ---------------------------------------------------------------- flash attention
// grid(qblk=8, h=3, b=32); block = 8 waves; each wave owns a 16-query tile.
// K/V 32-key chunks are DMA'd into double-buffered LDS by the Tensor Data
// Mover (wave 0 issues, TENSORcnt waits), overlapping DMA with WMMA math.
__global__ __launch_bounds__(256) void attn_kernel(const _Float16* __restrict__ Qg,
                                                   const _Float16* __restrict__ Kg,
                                                   const _Float16* __restrict__ Vg,
                                                   _Float16* __restrict__ Og) {
  __shared__ __align__(16) _Float16 kbuf[2][32 * 64];
  __shared__ __align__(16) _Float16 vbuf[2][32 * 64];
  __shared__ __align__(16) _Float16 pbuf[8][16 * 32];
  const int wv = threadIdx.x >> 5;
  const int lane = threadIdx.x & 31, lh = lane & 15, hi = lane >> 4;
  const int h = blockIdx.y;
  const int b = blockIdx.z;
  const int qrow = b * T_ + blockIdx.x * 128 + wv * 16;  // global token row
  const bool dma_wave = (threadIdx.x < 32);              // wave 0 drives the TDM

  const _Float16* kb_base = Kg + (size_t)b * T_ * D_ + h * HD_;
  const _Float16* vb_base = Vg + (size_t)b * T_ * D_ + h * HD_;

  v16h qa0 = load_a(Qg, D_, qrow, h * HD_);
  v16h qa1 = load_a(Qg, D_, qrow, h * HD_ + 32);

  v8f zero = {0.f, 0.f, 0.f, 0.f, 0.f, 0.f, 0.f, 0.f};
  v8f acc[4] = {zero, zero, zero, zero};
  float mrow[8], lrow[8];
#pragma unroll
  for (int r = 0; r < 8; ++r) {
    mrow[r] = -1e30f;
    lrow[r] = 0.f;
  }

  // prologue: DMA first K/V chunk
  if (dma_wave) {
    tdm_load_2d_f16(kb_base, &kbuf[0][0], 32, 64, D_);
    tdm_load_2d_f16(vb_base, &vbuf[0][0], 32, 64, D_);
    __builtin_amdgcn_s_wait_tensorcnt(0);
  }
  __syncthreads();

  int cur = 0;
  for (int kc = 0; kc < T_; kc += 32) {
    // async-prefetch next chunk into the other buffer while we compute
    if (dma_wave && (kc + 32 < T_)) {
      tdm_load_2d_f16(kb_base + (size_t)(kc + 32) * D_, &kbuf[cur ^ 1][0], 32, 64, D_);
      tdm_load_2d_f16(vb_base + (size_t)(kc + 32) * D_, &vbuf[cur ^ 1][0], 32, 64, D_);
    }

    // scores S(16q x 32k) = Q . K^T : kbuf is [key][hd] == B^T layout for load_bT
    const _Float16* kcur = &kbuf[cur][0];
    const _Float16* vcur = &vbuf[cur][0];
    v8f s0 = zero, s1 = zero;
    s0 = __builtin_amdgcn_wmma_f32_16x16x32_f16(false, qa0, false, load_bT(kcur, 64, 0, 0),
                                                (short)0, s0, false, false);
    s0 = __builtin_amdgcn_wmma_f32_16x16x32_f16(false, qa1, false, load_bT(kcur, 64, 0, 32),
                                                (short)0, s0, false, false);
    s1 = __builtin_amdgcn_wmma_f32_16x16x32_f16(false, qa0, false, load_bT(kcur, 64, 16, 0),
                                                (short)0, s1, false, false);
    s1 = __builtin_amdgcn_wmma_f32_16x16x32_f16(false, qa1, false, load_bT(kcur, 64, 16, 32),
                                                (short)0, s1, false, false);

    // online softmax: rows live across a 16-lane group (C layout)
    float fac[8];
#pragma unroll
    for (int r = 0; r < 8; ++r) {
      float a0 = s0[r] * 0.125f, a1v = s1[r] * 0.125f;
      float vm = fmaxf(a0, a1v);
#pragma unroll
      for (int m = 8; m >= 1; m >>= 1) vm = fmaxf(vm, __shfl_xor(vm, m, 16));
      float mn = fmaxf(mrow[r], vm);
      fac[r] = __expf(mrow[r] - mn);
      mrow[r] = mn;
      float e0 = __expf(a0 - mn), e1 = __expf(a1v - mn);
      float rs = e0 + e1;
#pragma unroll
      for (int m = 8; m >= 1; m >>= 1) rs += __shfl_xor(rs, m, 16);
      lrow[r] = lrow[r] * fac[r] + rs;
      // transpose P from C-layout into A-layout staging buffer (per-wave)
      pbuf[wv][(hi * 8 + r) * 32 + lh] = (_Float16)e0;
      pbuf[wv][(hi * 8 + r) * 32 + 16 + lh] = (_Float16)e1;
    }
    asm volatile("s_wait_dscnt 0" ::: "memory");  // wave-local LDS RAW fence
    v16h pa = load_a(&pbuf[wv][0], 32, 0, 0);

#pragma unroll
    for (int j = 0; j < 4; ++j)
#pragma unroll
      for (int r = 0; r < 8; ++r) acc[j][r] *= fac[r];
#pragma unroll
    for (int j = 0; j < 4; ++j) {
      v16h vbv = load_b(vcur, 64, 0, j * 16);
      acc[j] = __builtin_amdgcn_wmma_f32_16x16x32_f16(false, pa, false, vbv, (short)0,
                                                      acc[j], false, false);
    }

    // make the prefetched chunk visible to every wave before the next round
    if (dma_wave && (kc + 32 < T_)) __builtin_amdgcn_s_wait_tensorcnt(0);
    __syncthreads();
    cur ^= 1;
  }

#pragma unroll
  for (int r = 0; r < 8; ++r) {
    float inv = 1.f / lrow[r];
#pragma unroll
    for (int j = 0; j < 4; ++j)
      Og[(size_t)(qrow + hi * 8 + r) * D_ + h * HD_ + j * 16 + lh] = (_Float16)(acc[j][r] * inv);
  }
}

// ---------------------------------------------------------------- LayerNorm2 (f32 row -> f16)
__global__ __launch_bounds__(256) void ln_kernel(const float* __restrict__ xx,
                                                 const float* __restrict__ w,
                                                 const float* __restrict__ bb,
                                                 _Float16* __restrict__ out) {
  int wv = threadIdx.x >> 5, lane = threadIdx.x & 31;
  size_t row = (size_t)blockIdx.x * 8 + wv;
  const float* p = xx + row * D_;
  float v[6];
  float s = 0.f;
#pragma unroll
  for (int k = 0; k < 6; ++k) {
    v[k] = p[lane + 32 * k];
    s += v[k];
  }
  float mean = wsum(s) * (1.f / 192.f);
  float vs = 0.f;
#pragma unroll
  for (int k = 0; k < 6; ++k) {
    float d = v[k] - mean;
    vs += d * d;
  }
  float rstd = rsqrtf(wsum(vs) * (1.f / 192.f) + 1e-5f);
#pragma unroll
  for (int k = 0; k < 6; ++k) {
    int dd = lane + 32 * k;
    out[row * D_ + dd] = (_Float16)((v[k] - mean) * rstd * w[dd] + bb[dd]);
  }
}

// ---------------------------------------------------------------- final combine
// d_out already holds x_res; add y[b,t,:] * h_post[b,t,n]
__global__ __launch_bounds__(256) void final_kernel(float* __restrict__ out,
                                                    const float* __restrict__ y,
                                                    const float* __restrict__ hpost) {
  size_t idx = (size_t)blockIdx.x * 256 + threadIdx.x;
  size_t bt = idx / 768;
  int rem = (int)(idx - bt * 768);
  int nn = rem / 192, dd = rem - nn * 192;
  out[idx] += y[bt * 192 + dd] * hpost[bt * 4 + nn];
}

// ================================================================ launcher
extern "C" void kernel_launch(void* const* d_in, const int* in_sizes, int n_in,
                              void* d_out, int out_size, void* d_ws, size_t ws_size,
                              hipStream_t stream) {
  (void)in_sizes; (void)n_in; (void)out_size; (void)ws_size;
  const float* x = (const float*)d_in[0];
  const float* norm_w = (const float*)d_in[1];
  const float* phi_pre = (const float*)d_in[2];
  const float* phi_post = (const float*)d_in[3];
  const float* phi_res = (const float*)d_in[4];
  const float* b_pre = (const float*)d_in[5];
  const float* b_post = (const float*)d_in[6];
  const float* b_res = (const float*)d_in[7];
  const float* a_pre = (const float*)d_in[8];
  const float* a_post = (const float*)d_in[9];
  const float* a_res = (const float*)d_in[10];
  const float* ln1w = (const float*)d_in[11];
  const float* ln1b = (const float*)d_in[12];
  const float* Wq = (const float*)d_in[13];
  const float* Wk = (const float*)d_in[14];
  const float* Wv = (const float*)d_in[15];
  const float* Wo = (const float*)d_in[16];
  const float* ln2w = (const float*)d_in[17];
  const float* ln2b = (const float*)d_in[18];
  const float* W1 = (const float*)d_in[19];
  const float* W2 = (const float*)d_in[20];

  char* wsp = (char*)d_ws;
  size_t off = 0;
  auto take = [&](size_t bytes) -> void* {
    void* p = wsp + off;
    off += (bytes + 255) & ~(size_t)255;
    return p;
  };
  _Float16* wq_t = (_Float16*)take((size_t)D_ * D_ * 2);
  _Float16* wk_t = (_Float16*)take((size_t)D_ * D_ * 2);
  _Float16* wv_t = (_Float16*)take((size_t)D_ * D_ * 2);
  _Float16* wo_t = (_Float16*)take((size_t)D_ * D_ * 2);
  _Float16* w1_t = (_Float16*)take((size_t)D_ * MLP_ * 2);
  _Float16* w2_t = (_Float16*)take((size_t)MLP_ * D_ * 2);
  float* xin_f = (float*)take((size_t)MT * D_ * 4);
  _Float16* h1_h = (_Float16*)take((size_t)MT * D_ * 2);
  float* hpost_f = (float*)take((size_t)MT * 4 * 4);
  _Float16* q_h = (_Float16*)take((size_t)MT * D_ * 2);
  _Float16* k_h = (_Float16*)take((size_t)MT * D_ * 2);
  _Float16* v_h = (_Float16*)take((size_t)MT * D_ * 2);
  _Float16* o_h = (_Float16*)take((size_t)MT * D_ * 2);
  float* x1_f = (float*)take((size_t)MT * D_ * 4);
  _Float16* h2_h = (_Float16*)take((size_t)MT * D_ * 2);
  _Float16* a1_h = (_Float16*)take((size_t)MT * MLP_ * 2);
  float* y_f = (float*)take((size_t)MT * D_ * 4);

  // weights -> f16, transposed to [N][K]
  convert_t_kernel<<<144, 256, 0, stream>>>(Wq, wq_t, D_, D_);
  convert_t_kernel<<<144, 256, 0, stream>>>(Wk, wk_t, D_, D_);
  convert_t_kernel<<<144, 256, 0, stream>>>(Wv, wv_t, D_, D_);
  convert_t_kernel<<<144, 256, 0, stream>>>(Wo, wo_t, D_, D_);
  convert_t_kernel<<<576, 256, 0, stream>>>(W1, w1_t, D_, MLP_);
  convert_t_kernel<<<576, 256, 0, stream>>>(W2, w2_t, MLP_, D_);

  // gating stage: writes x_res into d_out, x_in (f32), LN1(x_in) f16, h_post
  gate_kernel<<<MT / 8, 256, 0, stream>>>(x, norm_w, phi_pre, phi_post, phi_res, b_pre,
                                          b_post, b_res, a_pre, a_post, a_res, ln1w, ln1b,
                                          (float*)d_out, xin_f, h1_h, hpost_f);

  dim3 g192(MT / 128, 3), g768(MT / 128, 12);
  gemm_kernel<EPI_F16><<<g192, 256, 0, stream>>>(h1_h, wq_t, q_h, nullptr, MT, D_, D_);
  gemm_kernel<EPI_F16><<<g192, 256, 0, stream>>>(h1_h, wk_t, k_h, nullptr, MT, D_, D_);
  gemm_kernel<EPI_F16><<<g192, 256, 0, stream>>>(h1_h, wv_t, v_h, nullptr, MT, D_, D_);

  attn_kernel<<<dim3(T_ / 128, HEADS, B_), 256, 0, stream>>>(q_h, k_h, v_h, o_h);

  gemm_kernel<EPI_RES><<<g192, 256, 0, stream>>>(o_h, wo_t, x1_f, xin_f, MT, D_, D_);
  ln_kernel<<<MT / 8, 256, 0, stream>>>(x1_f, ln2w, ln2b, h2_h);
  gemm_kernel<EPI_GELU><<<g768, 256, 0, stream>>>(h2_h, w1_t, a1_h, nullptr, MT, MLP_, D_);
  gemm_kernel<EPI_RES><<<g192, 256, 0, stream>>>(a1_h, w2_t, y_f, x1_f, MT, D_, MLP_);

  final_kernel<<<(MT * ND) / 256, 256, 0, stream>>>((float*)d_out, y_f, hpost_f);
}
